// KGATRecommender_40140764349010
// MI455X (gfx1250) — matrix operand order repaired and checked by reading, not compiled, not run on hardware
//
#include <hip/hip_runtime.h>
#include <hip/hip_bf16.h>
#include <math.h>

#define NUM_USERS 50000
#define NUM_ITEMS 50000
#define NNODES    100000
#define D         128
#define D2        256
#define NLAYERS   3

typedef __attribute__((ext_vector_type(16))) __bf16 v16bf;
typedef __attribute__((ext_vector_type(8)))  __bf16 v8bf;
typedef __attribute__((ext_vector_type(8)))  float  v8f;

// ---------------------------------------------------------------------------
// x0 = concat(user_emb, item_emb)  (float4 copies, coalesced)
// ---------------------------------------------------------------------------
__global__ void k_concat(const float* __restrict__ ue, const float* __restrict__ ie,
                         float* __restrict__ x) {
  long i = (long)blockIdx.x * blockDim.x + threadIdx.x;   // float4 index
  long n4 = (long)NNODES * D / 4;
  if (i >= n4) return;
  long e = i * 4;
  const long ub = (long)NUM_USERS * D;
  const float* s = (e < ub) ? (ue + e) : (ie + (e - ub));
  *(float4*)(x + e) = *(const float4*)s;
}

__global__ void k_zero(float* __restrict__ p, long n4) {
  long i = (long)blockIdx.x * blockDim.x + threadIdx.x;
  if (i < n4) *(float4*)(p + i * 4) = make_float4(0.f, 0.f, 0.f, 0.f);
}

// ---------------------------------------------------------------------------
// Edge phase: one wave32 per edge. lane holds 4 of 128 channels (float4).
// att = sigmoid(dot(x[src], Wa[:128]) + dot(x[dst], Wa[128:]) + ba)
// agg[dst] += att * x[src]   (f32 global atomics; L2-resident)
// ---------------------------------------------------------------------------
__global__ void k_edge_att(const float* __restrict__ x, const int* __restrict__ src,
                           const int* __restrict__ dst, const float* __restrict__ Wa,
                           const float* __restrict__ ba, float* __restrict__ agg,
                           int nedges) {
  int lane = threadIdx.x & 31;
  int wave = threadIdx.x >> 5;
  long e = (long)blockIdx.x * (blockDim.x >> 5) + wave;
  if (e >= nedges) return;
  int s = src[e], d = dst[e];
  float4 xs = *(const float4*)(x + (long)s * D + lane * 4);
  float4 xd = *(const float4*)(x + (long)d * D + lane * 4);
  float4 wa = *(const float4*)(Wa + lane * 4);
  float4 wb = *(const float4*)(Wa + D + lane * 4);
  float p = xs.x * wa.x + xs.y * wa.y + xs.z * wa.z + xs.w * wa.w
          + xd.x * wb.x + xd.y * wb.y + xd.z * wb.z + xd.w * wb.w;
#pragma unroll
  for (int off = 16; off > 0; off >>= 1) p += __shfl_xor(p, off, 32);
  float att = 1.0f / (1.0f + __expf(-(p + ba[0])));
  float* ap = agg + (long)d * D + lane * 4;
  atomicAdd(ap + 0, att * xs.x);
  atomicAdd(ap + 1, att * xs.y);
  atomicAdd(ap + 2, att * xs.z);
  atomicAdd(ap + 3, att * xs.w);
}

// ---------------------------------------------------------------------------
// Pre-fragment Wg[l] (f32 [256,128]) into bf16 WMMA B-lane layout:
// frag f = ki*8+nt; lane: N = nt*16 + (lane&15), K = ki*32 + (lane>>4)*8 + e + (e>=8 ? 8 : 0).
// Each GEMM lane then loads its v16bf B-fragment as one coalesced 32B read.
// ---------------------------------------------------------------------------
__global__ void k_prep_wg(const float* __restrict__ Wg, __bf16* __restrict__ frag) {
  int t = blockIdx.x * blockDim.x + threadIdx.x;   // (f, lane) pair
  if (t >= 64 * 32) return;
  int f = t >> 5, lane = t & 31;
  int ki = f >> 3, nt = f & 7;
  int n  = nt * 16 + (lane & 15);
  int kb = ki * 32 + (lane >> 4) * 8;
  __bf16* o = frag + (size_t)t * 16;
#pragma unroll
  for (int e = 0; e < 16; ++e) o[e] = (__bf16)Wg[(long)(kb + e + (e >= 8 ? 8 : 0)) * D + n];
}

// ---------------------------------------------------------------------------
// x_out = relu([x | agg] @ Wg + bg) via v_wmma_f32_16x16x32_bf16.
// Block = 128 threads (4 waves), 64-row tile. Wave -> 16x128 strip:
// 8 K-steps x 8 N-tiles = 64 WMMAs, fully unrolled.
// ---------------------------------------------------------------------------
#define ROWS_PER_BLOCK 64
#define LDA 264   // bf16 elems per padded LDS row; 264*2=528 bytes (16B aligned)

__global__ void __launch_bounds__(128)
k_gemm_relu(const float* __restrict__ x, const float* __restrict__ agg,
            const __bf16* __restrict__ wfrag, const float* __restrict__ bg,
            float* __restrict__ out) {
  __shared__ __bf16 lA[ROWS_PER_BLOCK * LDA];   // 33.8 KB of the 320 KB/WGP
  int rowbase = blockIdx.x * ROWS_PER_BLOCK;

  // Cooperative load of the A tile ([x | agg] rows), f32 -> bf16 into LDS.
  for (int i = threadIdx.x; i < ROWS_PER_BLOCK * D2; i += blockDim.x) {
    int r = i >> 8, c = i & 255;
    int m = rowbase + r;
    float v = 0.f;
    if (m < NNODES) v = (c < D) ? x[(long)m * D + c] : agg[(long)m * D + (c - D)];
    lA[r * LDA + c] = (__bf16)v;
  }
  __syncthreads();

  int lane = threadIdx.x & 31;
  int wave = threadIdx.x >> 5;
  int mrow = wave * 16 + (lane & 15);   // A-fragment row (ISA 16-bit A layout)
  int kb   = (lane >> 4) * 8;           // sub-K base within the 32-wide K chunk

  v8f acc[8];
#pragma unroll
  for (int nt = 0; nt < 8; ++nt)
    acc[nt] = (v8f){0.f, 0.f, 0.f, 0.f, 0.f, 0.f, 0.f, 0.f};

#pragma unroll
  for (int ki = 0; ki < 8; ++ki) {
    const __bf16* ap = &lA[mrow * LDA + ki * 32 + kb];
    v8bf lo = *(const v8bf*)ap;          // K = kb .. kb+7     (ds_load_b128)
    v8bf hi = *(const v8bf*)(ap + 16);   // K = kb+16 .. kb+23 (ds_load_b128)
    v16bf a = __builtin_shufflevector(lo, hi, 0, 1, 2, 3, 4, 5, 6, 7,
                                              8, 9, 10, 11, 12, 13, 14, 15);
#pragma unroll
    for (int nt = 0; nt < 8; ++nt) {
      v16bf b = *(const v16bf*)(wfrag + ((size_t)(ki * 8 + nt) * 32 + lane) * 16);
      acc[nt] = __builtin_amdgcn_wmma_f32_16x16x32_bf16(
          false, a, false, b, (short)0, acc[nt], false, false);
    }
  }

  // Epilogue: bias + ReLU, per C/D layout (VGPR r -> M=r / r+8, N=lane&15).
  int mbase = rowbase + wave * 16 + ((lane >> 4) ? 8 : 0);
  int col0  = lane & 15;
#pragma unroll
  for (int nt = 0; nt < 8; ++nt) {
    int col = nt * 16 + col0;
    float bb = bg[col];
#pragma unroll
    for (int r = 0; r < 8; ++r) {
      int m = mbase + r;
      if (m < NNODES) {
        float v = acc[nt][r] + bb;
        out[(long)m * D + col] = v > 0.f ? v : 0.f;
      }
    }
  }
}

// ---------------------------------------------------------------------------
extern "C" void kernel_launch(void* const* d_in, const int* in_sizes, int n_in,
                              void* d_out, int out_size, void* d_ws, size_t ws_size,
                              hipStream_t stream) {
  const int*   edge = (const int*)d_in[0];    // (2, E) int32
  const float* ue   = (const float*)d_in[1];  // (U, 128)
  const float* ie   = (const float*)d_in[2];  // (I, 128)
  const float* Wa   = (const float*)d_in[3];  // (3, 256, 1)
  const float* ba   = (const float*)d_in[4];  // (3, 1)
  const float* Wg   = (const float*)d_in[5];  // (3, 256, 128)
  const float* bg   = (const float*)d_in[6];  // (3, 128)

  const int E = in_sizes[0] / 2;
  const int* src = edge;
  const int* dst = edge + E;

  // Workspace layout: x0 | x1 | agg | wfrag   (~153.7 MB)
  float*  x0    = (float*)d_ws;
  float*  x1    = x0 + (size_t)NNODES * D;
  float*  agg   = x1 + (size_t)NNODES * D;
  __bf16* wfrag = (__bf16*)(agg + (size_t)NNODES * D);   // 64*32*16 bf16 = 64 KB

  long n4 = (long)NNODES * D / 4;
  k_concat<<<(int)((n4 + 255) / 256), 256, 0, stream>>>(ue, ie, x0);

  float* xin_seq[3] = {x0, x1, x0};   // ping-pong; last layer -> d_out
  for (int l = 0; l < NLAYERS; ++l) {
    float* xin  = xin_seq[l];
    float* xout = (l == NLAYERS - 1) ? (float*)d_out : xin_seq[l + 1];

    k_zero<<<(int)((n4 + 255) / 256), 256, 0, stream>>>(agg, n4);
    k_prep_wg<<<(64 * 32 + 127) / 128, 128, 0, stream>>>(Wg + (size_t)l * D2 * D, wfrag);

    const int epb = 256 / 32;   // 8 edges per 256-thread block (wave per edge)
    k_edge_att<<<(E + epb - 1) / epb, 256, 0, stream>>>(
        xin, src, dst, Wa + (size_t)l * D2, ba + l, agg, E);

    k_gemm_relu<<<(NNODES + ROWS_PER_BLOCK - 1) / ROWS_PER_BLOCK, 128, 0, stream>>>(
        xin, agg, wfrag, bg + (size_t)l * D, xout);
  }
}
